// LeafBlockAttention_69157563400522
// MI455X (gfx1250) — matrix operand order repaired and checked by reading, not compiled, not run on hardware
//
#include <hip/hip_runtime.h>
#include <hip/hip_bf16.h>

// ---------------------------------------------------------------------------
// Types / helpers
// ---------------------------------------------------------------------------
typedef __attribute__((ext_vector_type(16))) __bf16 v16bf;
typedef __attribute__((ext_vector_type(8)))  __bf16 v8bf;
typedef __attribute__((ext_vector_type(8)))  float  v8f;
typedef __attribute__((ext_vector_type(4)))  float  v4f;
typedef __attribute__((ext_vector_type(4)))  int    v4i;

#define CDIM 256
#define NH   4
#define HD   64
#define LDA  264   // x / scores scratch stride (f32)
#define LDQ2 520   // q|k stride (bf16)
#define LDX  264   // x_out stride (bf16)

#if __has_builtin(__builtin_amdgcn_global_load_async_to_lds_b128) && \
    __has_builtin(__builtin_amdgcn_s_wait_asynccnt)
#define USE_ASYNC_LDS 1
#else
#define USE_ASYNC_LDS 0
#endif

static __device__ inline v8f wmma_bf16(v16bf a, v16bf b, v8f c) {
  return __builtin_amdgcn_wmma_f32_16x16x32_bf16(false, a, false, b, (short)0, c,
                                                 false, false);
}

static __device__ inline float bfbits_to_f(unsigned int u) {
  union { unsigned int i; float f; } v; v.i = u << 16; return v.f;
}
static __device__ inline unsigned short f_to_bfbits(float f) {
  union { float f; unsigned int i; } v; v.f = f;
  unsigned int x = v.i + 0x7FFFu + ((v.i >> 16) & 1u);  // RNE
  return (unsigned short)(x >> 16);
}

// A fragment (16x32) from f32 LDS: lane L -> row M = L&15; K halves split by L>>4.
static __device__ inline v16bf load_a_f32v(const float* S, int ld, int mbase,
                                           int kbase, int mMax, int lane) {
  int m  = mbase + (lane & 15);
  int kb = (lane >> 4) * 8;
  v16bf a;
  if (m < mMax) {
    const float* p = S + m * ld + kbase + kb;
    v4f f0 = *(const v4f*)(p);
    v4f f1 = *(const v4f*)(p + 4);
    v4f f2 = *(const v4f*)(p + 16);
    v4f f3 = *(const v4f*)(p + 20);
#pragma unroll
    for (int e = 0; e < 4; ++e) {
      a[e]      = (__bf16)f0[e];
      a[4 + e]  = (__bf16)f1[e];
      a[8 + e]  = (__bf16)f2[e];
      a[12 + e] = (__bf16)f3[e];
    }
  } else {
#pragma unroll
    for (int e = 0; e < 16; ++e) a[e] = (__bf16)0.0f;
  }
  return a;
}

// A fragment (16x32) from bf16 LDS, 16-B vector loads.
static __device__ inline v16bf load_a_bf16(const __bf16* S, int ld, int mbase,
                                           int kbase, int lane) {
  int m  = mbase + (lane & 15);
  int kb = (lane >> 4) * 8;
  const __bf16* p = S + m * ld + kbase + kb;
  v8bf lo = *(const v8bf*)(p);
  v8bf hi = *(const v8bf*)(p + 16);
  return __builtin_shufflevector(lo, hi, 0, 1, 2, 3, 4, 5, 6, 7, 8, 9, 10, 11,
                                 12, 13, 14, 15);
}

// B fragment where B[k][n] = S[n*ld + colbase + k] (row n holds K contiguously).
static __device__ inline v16bf load_bT_lds(const __bf16* S, int ld, int colbase,
                                           int kbase, int nbase, int lane) {
  int n  = nbase + (lane & 15);
  int kb = kbase + ((lane >> 4) << 3);
  const __bf16* p = S + n * ld + colbase + kb;
  v8bf lo = *(const v8bf*)(p);
  v8bf hi = *(const v8bf*)(p + 16);
  return __builtin_shufflevector(lo, hi, 0, 1, 2, 3, 4, 5, 6, 7, 8, 9, 10, 11,
                                 12, 13, 14, 15);
}

// ---------------------------------------------------------------------------
// K0: pack f32 weight w[n][k] (n rows of 256) into fragment-ready bf16 tiles:
//     dst[((kt*NT + nt)*32 + lane)*16 + e] = w[(nt*16 + lane&15)*256 +
//                                              kt*32 + (lane>>4)*8 + (e&7) + (e>>3)*16]
// ---------------------------------------------------------------------------
__global__ void pack_w_kernel(const float* __restrict__ w,
                              __bf16* __restrict__ dst, int NT) {
  int i = blockIdx.x * blockDim.x + threadIdx.x;
  int total = 8 * NT * 512;
  if (i >= total) return;
  int e  = i & 15;
  int L  = (i >> 4) & 31;
  int tl = i >> 9;
  int nt = tl % NT;
  int kt = tl / NT;
  int n  = nt * 16 + (L & 15);
  int k  = kt * 32 + ((L >> 4) << 3) + (e & 7) + ((e >> 3) << 4);
  dst[i] = (__bf16)w[(size_t)n * 256 + k];
}

// ---------------------------------------------------------------------------
// K1: init workspace: zero edge-feature table, seed mask rows (diag + col 32)
// ---------------------------------------------------------------------------
__global__ void init_ws_kernel(unsigned long long* __restrict__ maskws,
                               unsigned long long* __restrict__ efws,
                               long nmask, long nef) {
  long stride = (long)gridDim.x * blockDim.x;
  long i0 = (long)blockIdx.x * blockDim.x + threadIdx.x;
  for (long i = i0; i < nef; i += stride) efws[i] = 0ull;
  for (long i = i0; i < nmask; i += stride)
    maskws[i] = (1ull << (i & 31)) | (1ull << 32);
}

// ---------------------------------------------------------------------------
// K2: scatter edges -> mask bits + packed bf16 edge features (dx, dy, dz, val)
// ---------------------------------------------------------------------------
__global__ void edge_scatter_kernel(const int* __restrict__ ei,
                                    const float* __restrict__ ev,
                                    const float* __restrict__ pos,
                                    unsigned long long* __restrict__ maskws,
                                    unsigned long long* __restrict__ efws,
                                    int nb, int E) {
  int e = blockIdx.x * blockDim.x + threadIdx.x;
  if (e >= E) return;
  int r = ei[e], c = ei[E + e];
  int br = r >> 5;
  if (br != (c >> 5) || br >= nb) return;
  int rl = r & 31, cl = c & 31;
  atomicOr(&maskws[(size_t)br * 32 + rl], 1ull << cl);
  float d0 = pos[3 * c + 0] - pos[3 * r + 0];
  float d1 = pos[3 * c + 1] - pos[3 * r + 1];
  float d2 = pos[3 * c + 2] - pos[3 * r + 2];
  float v3 = ev[e];
  unsigned long long p = (unsigned long long)f_to_bfbits(d0)
                       | ((unsigned long long)f_to_bfbits(d1) << 16)
                       | ((unsigned long long)f_to_bfbits(d2) << 32)
                       | ((unsigned long long)f_to_bfbits(v3) << 48);
  efws[((size_t)br * 32 + rl) * 33 + cl] = p;
}

// ---------------------------------------------------------------------------
// K3: fused per-block attention. 1 workgroup (8 wave32) per 32-node block.
// ---------------------------------------------------------------------------
__global__ __launch_bounds__(256)
void block_attn_kernel(const float* __restrict__ x,
                       const unsigned long long* __restrict__ maskws,
                       const unsigned long long* __restrict__ efws,
                       const __bf16* __restrict__ W2,   // packed qkv frags
                       const float* __restrict__ qkv_b, // [768]
                       const __bf16* __restrict__ Wp,   // packed proj frags
                       const float* __restrict__ proj_b,// [256]
                       const float* __restrict__ gate_w,// [4][4]
                       const float* __restrict__ gate_b,// [4]
                       float* __restrict__ out, int nb) {
  // x+mean f32 (34,848 B) reused as scores f32 (24,576 B)
  __shared__ __attribute__((aligned(16))) char  sScratchRaw[33 * LDA * 4];
  // q|k bf16, rows 0..47 (49,920 B); reused as f32 out staging (32,768 B)
  __shared__ __attribute__((aligned(16))) char  sQKraw[48 * LDQ2 * 2];
  __shared__ __attribute__((aligned(16))) __bf16 sVT[256 * 64];     // v^T, 32,768 B
  __shared__ __attribute__((aligned(16))) __bf16 sComb[NH * 32 * 64]; // 16,384 B
  __shared__ __attribute__((aligned(16))) __bf16 sXout[32 * LDX];     // 16,896 B

  float*  sA  = (float*)sScratchRaw;
  __bf16* sQK = (__bf16*)sQKraw;
  const v16bf* W2f = (const v16bf*)W2;
  const v16bf* Wpf = (const v16bf*)Wp;

  const int b    = blockIdx.x;
  const int tid  = threadIdx.x;
  const int w    = tid >> 5;
  const int lane = tid & 31;

  // ---- stage 0: async-stage x block into LDS; zero sVT pad cols ----
  const float* xb = x + (size_t)b * 32 * CDIM;
#if USE_ASYNC_LDS
  for (int i = tid; i < 2048; i += 256) {
    int r = i >> 6, q = i & 63;
    __builtin_amdgcn_global_load_async_to_lds_b128(
        (__attribute__((address_space(1))) v4i*)(xb + r * CDIM + q * 4),
        (__attribute__((address_space(3))) v4i*)(sA + r * LDA + q * 4), 0, 0);
  }
#else
  for (int i = tid; i < 32 * CDIM; i += 256) {
    int r = i >> 8, c = i & 255;
    sA[r * LDA + c] = xb[i];
  }
#endif
  for (int i = tid; i < 256 * 16; i += 256) {
    int d = i >> 4, c = 48 + (i & 15);
    sVT[d * 64 + c] = (__bf16)0.0f;
  }
#if USE_ASYNC_LDS
  __builtin_amdgcn_s_wait_asynccnt(0);
#endif
  __syncthreads();

  if (tid < CDIM) {
    float s = 0.f;
#pragma unroll
    for (int r = 0; r < 32; ++r) s += sA[r * LDA + tid];
    sA[32 * LDA + tid] = s * (1.0f / 32.0f);
  }
  __syncthreads();

  // ---- stage 1: QKV GEMM (48x256)x(256x768); q,k row-major; v transposed ----
  for (int mt = 0; mt < 3; ++mt) {
    v16bf afr[8];
#pragma unroll
    for (int kt = 0; kt < 8; ++kt)
      afr[kt] = load_a_f32v(sA, LDA, mt * 16, kt * 32, 33, lane);
    for (int nt = w; nt < 48; nt += 8) {
      v8f acc = {};
#pragma unroll
      for (int kt = 0; kt < 8; ++kt)
        acc = wmma_bf16(afr[kt], W2f[(kt * 48 + nt) * 32 + lane], acc);
      int n    = nt * 16 + (lane & 15);
      int rb   = mt * 16 + ((lane >> 4) << 3);
      float bi = qkv_b[n];
      if (nt < 32) {  // q | k -> row-major
#pragma unroll
        for (int j = 0; j < 8; ++j)
          sQK[(rb + j) * LDQ2 + n] = (__bf16)(acc[j] + bi);
      } else {        // v -> transposed [d][kvrow]
        int d = n - 512;
        v8bf vv;
#pragma unroll
        for (int j = 0; j < 8; ++j) vv[j] = (__bf16)(acc[j] + bi);
        *(v8bf*)(sVT + d * 64 + rb) = vv;
      }
    }
  }
  __syncthreads();

  // ---- stage 2: scores = q . k^T per head (32x33, padded N=48) ----
  float* sSc = (float*)sScratchRaw;  // overwrite x
  for (int t = w; t < NH * 2 * 3; t += 8) {
    int h = t / 6, rem = t % 6, mt = rem / 3, nt = rem % 3;
    v8f acc = {};
#pragma unroll
    for (int kt = 0; kt < 2; ++kt) {
      v16bf a  = load_a_bf16(sQK, LDQ2, mt * 16, h * HD + kt * 32, lane);      // q
      v16bf bb = load_bT_lds(sQK, LDQ2, CDIM + h * HD, kt * 32, nt * 16, lane); // k^T
      acc = wmma_bf16(a, bb, acc);
    }
    int c  = nt * 16 + (lane & 15);
    int rb = mt * 16 + ((lane >> 4) << 3);
#pragma unroll
    for (int j = 0; j < 8; ++j)
      sSc[((h * 32) + rb + j) * 48 + c] = acc[j] * 0.125f;  // hd^-0.5
  }
  __syncthreads();

  // ---- stage 3: mask + physics bias + softmax + gate -> sComb (bf16) ----
  if (tid < NH * 32) {
    int h = tid >> 5, r = tid & 31;
    unsigned long long mbits = maskws[(size_t)b * 32 + r];
    const unsigned long long* efrow = efws + ((size_t)b * 32 + r) * 33;
    float gw0 = gate_w[h * 4 + 0], gw1 = gate_w[h * 4 + 1];
    float gw2 = gate_w[h * 4 + 2], gw3 = gate_w[h * 4 + 3];
    float gb  = gate_b[h];
    float ex[33];
    float mx = -3.0e38f;
#pragma unroll 1
    for (int c = 0; c < 33; ++c) {
      bool on = (mbits >> c) & 1ull;
      float b3;
      if (c == 32 || c == r) b3 = 1.0f;
      else b3 = bfbits_to_f((unsigned int)(efrow[c] >> 48) & 0xFFFFu);
      float s = on ? (sSc[((h * 32) + r) * 48 + c] + b3) : -INFINITY;
      ex[c] = s;
      mx = fmaxf(mx, s);
    }
    float sum = 0.f;
#pragma unroll 1
    for (int c = 0; c < 33; ++c) {
      float e = __expf(ex[c] - mx);
      ex[c] = e;
      sum += e;
    }
    float inv = 1.0f / sum;
    __bf16* crow = sComb + ((h * 32) + r) * 64;
#pragma unroll 1
    for (int c = 0; c < 33; ++c) {
      bool on = (mbits >> c) & 1ull;
      float g = 0.f;
      if (on) {
        float e0, e1, e2, e3;
        if (c == 32 || c == r) { e0 = e1 = e2 = 0.f; e3 = 1.f; }
        else {
          unsigned long long p = efrow[c];
          e0 = bfbits_to_f((unsigned int)(p      ) & 0xFFFFu);
          e1 = bfbits_to_f((unsigned int)(p >> 16) & 0xFFFFu);
          e2 = bfbits_to_f((unsigned int)(p >> 32) & 0xFFFFu);
          e3 = bfbits_to_f((unsigned int)(p >> 48) & 0xFFFFu);
        }
        g = e0 * gw0 + e1 * gw1 + e2 * gw2 + e3 * gw3 + gb;
      }
      crow[c] = (__bf16)(ex[c] * inv + g);
    }
#pragma unroll
    for (int c = 33; c < 64; ++c) crow[c] = (__bf16)0.0f;
  }
  __syncthreads();

  // ---- stage 4: x_out = combined @ v (per head 32x64, K padded to 64) ----
  for (int t = w; t < NH * 2 * 4; t += 8) {
    int h = t >> 3, mt = (t >> 2) & 1, nt = t & 3;
    v8f acc = {};
#pragma unroll
    for (int kt = 0; kt < 2; ++kt) {
      v16bf a  = load_a_bf16(sComb + h * 32 * 64, 64, mt * 16, kt * 32, lane);
      // B[k=c][n=d] = sVT[(h*64+d)*64 + c]  (contiguous in c)
      v16bf bb = load_bT_lds(sVT, 64, 0, kt * 32, h * HD + nt * 16, lane);
      acc = wmma_bf16(a, bb, acc);
    }
    int d  = nt * 16 + (lane & 15);
    int rb = mt * 16 + ((lane >> 4) << 3);
#pragma unroll
    for (int j = 0; j < 8; ++j)
      sXout[(rb + j) * LDX + h * HD + d] = (__bf16)acc[j];
  }
  __syncthreads();

  // ---- stage 5: out = x_out @ proj^T + proj_b; stage f32 in LDS ----
  float* sOutF = (float*)sQKraw;  // 32KB, QK dead now
  for (int mt = 0; mt < 2; ++mt) {
    v16bf afr[8];
#pragma unroll
    for (int kt = 0; kt < 8; ++kt)
      afr[kt] = load_a_bf16(sXout, LDX, mt * 16, kt * 32, lane);
    for (int nt = w; nt < 16; nt += 8) {
      v8f acc = {};
#pragma unroll
      for (int kt = 0; kt < 8; ++kt)
        acc = wmma_bf16(afr[kt], Wpf[(kt * 16 + nt) * 32 + lane], acc);
      int n    = nt * 16 + (lane & 15);
      int rb   = mt * 16 + ((lane >> 4) << 3);
      float pb = proj_b[n];
#pragma unroll
      for (int j = 0; j < 8; ++j)
        sOutF[(rb + j) * CDIM + n] = acc[j] + pb;
    }
  }
  __syncthreads();

  // coalesced vectorized writeback
  float* outb = out + (size_t)b * 32 * CDIM;
  for (int i = tid; i < (32 * CDIM) / 4; i += 256)
    ((v4f*)outb)[i] = ((const v4f*)sOutF)[i];
}

// ---------------------------------------------------------------------------
// Launch
// ---------------------------------------------------------------------------
extern "C" void kernel_launch(void* const* d_in, const int* in_sizes, int n_in,
                              void* d_out, int out_size, void* d_ws, size_t ws_size,
                              hipStream_t stream) {
  const float* x      = (const float*)d_in[0];
  const int*   ei     = (const int*)d_in[1];
  const float* ev     = (const float*)d_in[2];
  const float* pos    = (const float*)d_in[3];
  const float* qkv_w  = (const float*)d_in[4];
  const float* qkv_b  = (const float*)d_in[5];
  const float* proj_w = (const float*)d_in[6];
  const float* proj_b = (const float*)d_in[7];
  const float* gate_w = (const float*)d_in[8];
  const float* gate_b = (const float*)d_in[9];
  float* out = (float*)d_out;

  const int N  = in_sizes[0] / CDIM;  // B=1
  const int nb = N / 32;
  const int E  = in_sizes[2];

  // workspace layout
  char* ws = (char*)d_ws;
  __bf16* W2 = (__bf16*)ws;                                   // 8*48*512 bf16
  __bf16* Wp = (__bf16*)(ws + 393216);                        // 8*16*512 bf16
  unsigned long long* maskws = (unsigned long long*)(ws + 524288);
  unsigned long long* efws =
      (unsigned long long*)(ws + 524288 + (size_t)nb * 32 * 8);

  pack_w_kernel<<<(8 * 48 * 512 + 255) / 256, 256, 0, stream>>>(qkv_w, W2, 48);
  pack_w_kernel<<<(8 * 16 * 512 + 255) / 256, 256, 0, stream>>>(proj_w, Wp, 16);

  long nmask = (long)nb * 32;
  long nef   = nmask * 33;
  init_ws_kernel<<<2048, 256, 0, stream>>>(maskws, efws, nmask, nef);
  edge_scatter_kernel<<<(E + 255) / 256, 256, 0, stream>>>(ei, ev, pos, maskws,
                                                           efws, nb, E);
  block_attn_kernel<<<nb, 256, 0, stream>>>(x, maskws, efws, W2, qkv_b, Wp,
                                            proj_b, gate_w, gate_b, out, nb);
}
